// GRUModule_32985348833742
// MI455X (gfx1250) — matrix operand order repaired and checked by reading, not compiled, not run on hardware
//
#include <hip/hip_runtime.h>
#include <hip/hip_bf16.h>

typedef __attribute__((ext_vector_type(16))) _Float16 v16h;
typedef __attribute__((ext_vector_type(8)))  _Float16 v8h;
typedef __attribute__((ext_vector_type(8)))  float    v8f;

#define T_STEPS 256
#define BATCH   256
#define HID     512
#define G3      1536   // 3*HID
#define BM      16     // batch rows per scan workgroup
#define HROW    516    // padded LDS row stride (floats) for scan h
#define BPAD    40     // padded LDS row stride (halfs) for gi_gemm B slice (80B)

// ---------------------------------------------------------------------------
// CDNA5 async global->LDS copy (ASYNCcnt-tracked), per-lane 16B.
// ldsoff = wave-relative LDS byte address (low 32 bits of generic pointer).
// ---------------------------------------------------------------------------
__device__ inline void async_ld_b128(unsigned ldsoff, const void* gaddr) {
    asm volatile("global_load_async_to_lds_b128 %0, %1, off"
                 :: "v"(ldsoff), "v"((unsigned long long)(size_t)gaddr)
                 : "memory");
}
__device__ inline void wait_async_le2() {
    asm volatile("s_wait_asynccnt 0x2" ::: "memory");
}
__device__ inline void wait_async_0() {
    asm volatile("s_wait_asynccnt 0x0" ::: "memory");
}

// ---------------------------------------------------------------------------
// f32 -> f16 weight conversion
// ---------------------------------------------------------------------------
__global__ void cvt_w_kernel(const float* __restrict__ a, _Float16* __restrict__ o, int n) {
    int i = blockIdx.x * blockDim.x + threadIdx.x;
    if (i < n) o[i] = (_Float16)a[i];
}

// B-matrix fragment from global: W row-major (N,K) f16; 16 contiguous halfs/lane.
__device__ inline v16h load_b_frag(const _Float16* __restrict__ W, int n0, int lane, int k0) {
    int n    = n0 + (lane & 15);
    int koff = (lane >> 4) * 16;
    return *(const v16h*)(W + (size_t)n * 512 + k0 + koff);
}

// A-matrix fragment from f32 row pointer (already offset to row*ld + k0 + half*16).
__device__ inline v16h make_a_frag(const float* __restrict__ p0) {
    v16h a;
#pragma unroll
    for (int i = 0; i < 8; ++i) {
        a[i]     = (_Float16)p0[i];
        a[8 + i] = (_Float16)p0[8 + i];
    }
    return a;
}

// ---------------------------------------------------------------------------
// gi = x @ W_ih^T + b_ih : (T*B,512) x (1536,512) -> (T*B,1536) f32
// Block = 8 waves = 128(M) x 128(N); each wave one 16x128 strip.
// B k-slice (128N x 32K f16 = 8KB) staged in LDS via async copies,
// double-buffered: waves read B via ds_load, next slice in flight.
// ---------------------------------------------------------------------------
__global__ __launch_bounds__(256) void gi_gemm_kernel(
    const float* __restrict__ x, const _Float16* __restrict__ Wih,
    const float* __restrict__ b_ih, float* __restrict__ gi)
{
    __shared__ _Float16 Bsh[2][128 * BPAD];   // 2 x 10240B

    int tid  = threadIdx.x;
    int lane = tid & 31;
    int w    = tid >> 5;
    int mb   = blockIdx.x / 12;     // 0..511  (65536/128 M-blocks)
    int ns   = blockIdx.x % 12;     // 0..11   (1536/128 N-strips)
    int m0   = mb * 128 + w * 16;   // this wave's M-tile
    int n0   = ns * 128;
    int half = lane >> 4;
    int mrow = m0 + (lane & 15);

    // per-lane chunk coordinates for staging (2 async ops per wave per slice)
    // chunk = w*64 + q*32 + lane, q in {0,1}; row = chunk>>2, quarter = chunk&3
    int  ch0   = w * 64 + lane;
    int  row0  = ch0 >> 2, q0 = ch0 & 3;
    int  ch1   = ch0 + 32;
    int  row1  = ch1 >> 2, q1 = ch1 & 3;
    const _Float16* g0 = Wih + (size_t)(n0 + row0) * 512 + q0 * 8;
    const _Float16* g1 = Wih + (size_t)(n0 + row1) * 512 + q1 * 8;
    unsigned l0a = (unsigned)(size_t)(const void*)&Bsh[0][row0 * BPAD + q0 * 8];
    unsigned l1a = (unsigned)(size_t)(const void*)&Bsh[0][row1 * BPAD + q1 * 8];
    unsigned l0b = (unsigned)(size_t)(const void*)&Bsh[1][row0 * BPAD + q0 * 8];
    unsigned l1b = (unsigned)(size_t)(const void*)&Bsh[1][row1 * BPAD + q1 * 8];

    v8f acc[8];
#pragma unroll
    for (int j = 0; j < 8; ++j)
#pragma unroll
        for (int r = 0; r < 8; ++r) acc[j][r] = 0.0f;

    // prime buffer 0 with k-slice 0
    async_ld_b128(l0a, g0);
    async_ld_b128(l1a, g1);

    for (int kk = 0; kk < 16; ++kk) {
        int k0 = kk * 32;
        __syncthreads();                      // everyone done reading the buffer we refill
        if (kk + 1 < 16) {                    // issue next slice into other buffer
            int knext = (kk + 1) * 32;
            if ((kk + 1) & 1) {
                async_ld_b128(l0b, g0 + knext);
                async_ld_b128(l1b, g1 + knext);
            } else {
                async_ld_b128(l0a, g0 + knext);
                async_ld_b128(l1a, g1 + knext);
            }
            wait_async_le2();                 // slice kk complete (in-order), kk+1 may fly
        } else {
            wait_async_0();
        }
        __syncthreads();                      // slice kk visible to all waves

        const _Float16* bs = Bsh[kk & 1];
        v16h a = make_a_frag(x + (size_t)mrow * 512 + k0 + half * 16);
#pragma unroll
        for (int j = 0; j < 8; ++j) {
            int  idx = (j * 16 + (lane & 15)) * BPAD + (lane >> 4) * 16;
            v8h  lo  = *(const v8h*)&bs[idx];
            v8h  hi  = *(const v8h*)&bs[idx + 8];
            v16h b;
#pragma unroll
            for (int i = 0; i < 8; ++i) { b[i] = lo[i]; b[8 + i] = hi[i]; }
            acc[j] = __builtin_amdgcn_wmma_f32_16x16x32_f16(
                false, a, false, b, (short)0, acc[j], false, false);
        }
    }

#pragma unroll
    for (int j = 0; j < 8; ++j) {
        int   n    = n0 + j * 16 + (lane & 15);
        float bias = b_ih[n];
#pragma unroll
        for (int r = 0; r < 8; ++r) {
            int m = m0 + half * 8 + r;
            gi[(size_t)m * G3 + n] = acc[j][r] + bias;
        }
    }
}

// ---------------------------------------------------------------------------
// Persistent masked-GRU scan + fused LayerNorm (unchanged structure).
// ---------------------------------------------------------------------------
__global__ __launch_bounds__(256) void gru_scan_kernel(
    const float* __restrict__ gi, const _Float16* __restrict__ Whh,
    const float* __restrict__ b_hh, const float* __restrict__ masks,
    const float* __restrict__ h0, const float* __restrict__ ln_w,
    const float* __restrict__ ln_b, float* __restrict__ y, float* __restrict__ hT)
{
    __shared__ float hs[BM * HROW];

    int tid  = threadIdx.x;
    int lane = tid & 31;
    int w    = tid >> 5;
    int half = lane >> 4;
    int b0   = blockIdx.x * BM;
    int cb   = w * 64;

    for (int i = tid; i < BM * HID; i += 256) {
        int r = i >> 9, c = i & 511;
        hs[r * HROW + c] = h0[(size_t)(b0 + r) * HID + c];
    }
    __syncthreads();

    for (int t = 0; t < T_STEPS; ++t) {
        for (int i = tid; i < BM * HID; i += 256) {
            int r = i >> 9, c = i & 511;
            hs[r * HROW + c] *= masks[(size_t)t * BATCH + b0 + r];
        }
        __syncthreads();

        size_t gbase = ((size_t)t * BATCH + b0) * G3;

        v8f aR[4], aZ[4], aN[4];
#pragma unroll
        for (int j = 0; j < 4; ++j) {
            int   c   = cb + j * 16 + (lane & 15);
            float bhr = b_hh[c], bhz = b_hh[HID + c], bhn = b_hh[2 * HID + c];
#pragma unroll
            for (int r = 0; r < 8; ++r) {
                int          m = half * 8 + r;
                const float* g = gi + gbase + (size_t)m * G3;
                aR[j][r] = g[c] + bhr;
                aZ[j][r] = g[HID + c] + bhz;
                aN[j][r] = bhn;
            }
        }

        for (int k0 = 0; k0 < 512; k0 += 32) {
            v16h a = make_a_frag(&hs[(lane & 15) * HROW + k0 + half * 16]);
#pragma unroll
            for (int j = 0; j < 4; ++j) {
                v16h br = load_b_frag(Whh, cb + j * 16, lane, k0);
                aR[j] = __builtin_amdgcn_wmma_f32_16x16x32_f16(
                    false, a, false, br, (short)0, aR[j], false, false);
                v16h bz = load_b_frag(Whh, HID + cb + j * 16, lane, k0);
                aZ[j] = __builtin_amdgcn_wmma_f32_16x16x32_f16(
                    false, a, false, bz, (short)0, aZ[j], false, false);
                v16h bn = load_b_frag(Whh, 2 * HID + cb + j * 16, lane, k0);
                aN[j] = __builtin_amdgcn_wmma_f32_16x16x32_f16(
                    false, a, false, bn, (short)0, aN[j], false, false);
            }
        }

        float hnew[4][8];
#pragma unroll
        for (int j = 0; j < 4; ++j) {
            int c = cb + j * 16 + (lane & 15);
#pragma unroll
            for (int r = 0; r < 8; ++r) {
                int   m    = half * 8 + r;
                float rr   = 1.0f / (1.0f + __expf(-aR[j][r]));
                float zz   = 1.0f / (1.0f + __expf(-aZ[j][r]));
                float gin  = gi[gbase + (size_t)m * G3 + 2 * HID + c];
                float nn   = tanhf(gin + rr * aN[j][r]);
                float hold = hs[m * HROW + c];
                hnew[j][r] = (1.0f - zz) * nn + zz * hold;
            }
        }
        __syncthreads();

#pragma unroll
        for (int j = 0; j < 4; ++j) {
            int c = cb + j * 16 + (lane & 15);
#pragma unroll
            for (int r = 0; r < 8; ++r)
                hs[(half * 8 + r) * HROW + c] = hnew[j][r];
        }
        __syncthreads();

#pragma unroll
        for (int rr = 0; rr < 2; ++rr) {
            int   row = w * 2 + rr;
            float s = 0.0f, s2 = 0.0f;
#pragma unroll
            for (int i = 0; i < 16; ++i) {
                float v = hs[row * HROW + lane + i * 32];
                s += v; s2 += v * v;
            }
#pragma unroll
            for (int off = 16; off > 0; off >>= 1) {
                s  += __shfl_xor(s, off, 32);
                s2 += __shfl_xor(s2, off, 32);
            }
            float  mu  = s * (1.0f / 512.0f);
            float  var = s2 * (1.0f / 512.0f) - mu * mu;
            float  inv = rsqrtf(var + 1e-5f);
            float* yr  = y + ((size_t)t * BATCH + b0 + row) * HID;
#pragma unroll
            for (int i = 0; i < 16; ++i) {
                int c2 = lane + i * 32;
                yr[c2] = (hs[row * HROW + c2] - mu) * inv * ln_w[c2] + ln_b[c2];
            }
        }
        __syncthreads();
    }

    for (int i = tid; i < BM * HID; i += 256) {
        int r = i >> 9, c = i & 511;
        hT[(size_t)(b0 + r) * HID + c] = hs[r * HROW + c];
    }
}

// ---------------------------------------------------------------------------
extern "C" void kernel_launch(void* const* d_in, const int* in_sizes, int n_in,
                              void* d_out, int out_size, void* d_ws, size_t ws_size,
                              hipStream_t stream) {
    const float* x     = (const float*)d_in[0];   // (T,B,I)
    const float* h0    = (const float*)d_in[1];   // (1,B,H)
    const float* masks = (const float*)d_in[2];   // (T,B,1)
    const float* Wih   = (const float*)d_in[3];   // (3H,I)
    const float* Whh   = (const float*)d_in[4];   // (3H,H)
    const float* bih   = (const float*)d_in[5];
    const float* bhh   = (const float*)d_in[6];
    const float* lnw   = (const float*)d_in[7];
    const float* lnb   = (const float*)d_in[8];

    float* y  = (float*)d_out;
    float* hT = y + (size_t)T_STEPS * BATCH * HID;

    char*      ws   = (char*)d_ws;
    _Float16*  WihH = (_Float16*)ws;
    _Float16*  WhhH = (_Float16*)(ws + (size_t)G3 * 512 * 2);
    float*     gi   = (float*)(ws + (size_t)2 * G3 * 512 * 2);

    const int WN = G3 * 512;
    cvt_w_kernel<<<(WN + 255) / 256, 256, 0, stream>>>(Wih, WihH, WN);
    cvt_w_kernel<<<(WN + 255) / 256, 256, 0, stream>>>(Whh, WhhH, WN);

    // 512 M-blocks (128 rows) * 12 N-strips
    gi_gemm_kernel<<<6144, 256, 0, stream>>>(x, WihH, bih, gi);

    gru_scan_kernel<<<BATCH / BM, 256, 0, stream>>>(gi, WhhH, bhh, masks, h0,
                                                    lnw, lnb, y, hT);
}
